// SS2D_48962627174855
// MI455X (gfx1250) — matrix operand order repaired and checked by reading, not compiled, not run on hardware
//
#include <hip/hip_runtime.h>
#include <hip/hip_bf16.h>

// ---------------------------------------------------------------------------
// SS2D fused pipeline for gfx1250 (MI455X), wave32 + WMMA bf16.
// Live dataflow only: in_proj GEMM -> dwconv3x3+SiLU -> Ds-scale * silu(z)
// -> LayerNorm -> out_proj GEMM.  (x_proj / dt / A_logs are dead code in the
// reference; cross_scan+merge collapse to a per-channel Ds scale.)
// This revision adds explicit register double-buffering in the WMMA K-loop so
// next-step fragment loads overlap the current WMMA group.
// ---------------------------------------------------------------------------

typedef __attribute__((ext_vector_type(16))) __bf16 v16bf;
typedef __attribute__((ext_vector_type(8)))  __bf16 v8bf;
typedef __attribute__((ext_vector_type(8)))  float  v8f;

#define HH 64
#define WW 64
#define D_INNER 384
#define D_MODEL 192
#define M_TOTAL 32768  // 8 * 64 * 64 pixels

static __device__ __forceinline__ __bf16 f2bf(float f) {
    union { float f; unsigned u; } in; in.f = f;
    unsigned u = in.u;
    unsigned r = u + 0x7FFFu + ((u >> 16) & 1u);   // round-to-nearest-even
    union { unsigned short s; __bf16 b; } out;
    out.s = (unsigned short)(r >> 16);
    return out.b;
}

// ---------------------------------------------------------------------------
// f32 -> bf16 conversion
// ---------------------------------------------------------------------------
__global__ void cvt_f32_bf16(const float* __restrict__ in, __bf16* __restrict__ out, int n) {
    int i = blockIdx.x * blockDim.x + threadIdx.x;
    if (i < n) out[i] = f2bf(in[i]);
}

// ---------------------------------------------------------------------------
// WMMA GEMM:  C[M x N] = A[M x KDIM] * Bt[N x KDIM]^T   (both bf16 row-major,
// Bt is the weight matrix as stored: out-features x in-features).
// Block: 256 threads = 8 waves; wave tile = 16(M) x 64(N); K step = 32,
// software-pipelined (double-buffered fragments).
// grid.x = M/128, grid.y = N/64.
// SPLIT epilogue (GEMM1): col<384 -> C0 raw (xh), col>=384 -> C1 = silu (zs).
// ---------------------------------------------------------------------------
template <int KDIM, int LDC, bool SPLIT>
__global__ __launch_bounds__(256)
void wmma_gemm_bt(const __bf16* __restrict__ A, const __bf16* __restrict__ Bt,
                  float* __restrict__ C0, float* __restrict__ C1) {
    const int lane  = threadIdx.x & 31;
    const int wave  = threadIdx.x >> 5;
    const int mBase = blockIdx.x * 128 + wave * 16;
    const int nBase = blockIdx.y * 64;

    // A fragment addressing (16-bit A 16x32 layout, contiguous K to match B):
    //   lanes 0-15: M=lane,    K 0..15 contiguous
    //   lanes16-31: M=lane-16, K 16..31 contiguous
    const int mRow = mBase + (lane & 15);
    const int aK   = (lane >> 4) * 16;
    // B fragment addressing (16-bit B 32x16 layout):
    //   lanes 0-15: N=lane,    K 0..15 contiguous; lanes16-31: N=lane-16, K 16..31
    const int nRow = lane & 15;
    const int bK   = (lane >> 4) * 16;

    const __bf16* __restrict__ ap = A + (size_t)mRow * KDIM + aK;
    const __bf16* __restrict__ bp0 = Bt + (size_t)(nBase + 0 * 16 + nRow) * KDIM + bK;
    const __bf16* __restrict__ bp1 = Bt + (size_t)(nBase + 1 * 16 + nRow) * KDIM + bK;
    const __bf16* __restrict__ bp2 = Bt + (size_t)(nBase + 2 * 16 + nRow) * KDIM + bK;
    const __bf16* __restrict__ bp3 = Bt + (size_t)(nBase + 3 * 16 + nRow) * KDIM + bK;

    auto loadA = [&](int k) -> v16bf {
        v8bf lo = *(const v8bf*)(ap + k);
        v8bf hi = *(const v8bf*)(ap + k + 8);
        v16bf r;
#pragma unroll
        for (int i = 0; i < 8; ++i) { r[i] = lo[i]; r[i + 8] = hi[i]; }
        return r;
    };

    v8f acc[4];
#pragma unroll
    for (int f = 0; f < 4; ++f)
#pragma unroll
        for (int i = 0; i < 8; ++i) acc[f][i] = 0.0f;

    // ---- software pipeline: prologue loads K-step 0 ----
    v16bf aC = loadA(0);
    v16bf bC[4];
    bC[0] = *(const v16bf*)(bp0);
    bC[1] = *(const v16bf*)(bp1);
    bC[2] = *(const v16bf*)(bp2);
    bC[3] = *(const v16bf*)(bp3);

#pragma unroll
    for (int k0 = 0; k0 < KDIM; k0 += 32) {
        const bool more = (k0 + 32) < KDIM;
        v16bf aN;
        v16bf bN[4];
        if (more) {                       // prefetch next K-step before WMMAs
            const int kn = k0 + 32;
            aN    = loadA(kn);
            bN[0] = *(const v16bf*)(bp0 + kn);
            bN[1] = *(const v16bf*)(bp1 + kn);
            bN[2] = *(const v16bf*)(bp2 + kn);
            bN[3] = *(const v16bf*)(bp3 + kn);
        }
#pragma unroll
        for (int f = 0; f < 4; ++f) {
            acc[f] = __builtin_amdgcn_wmma_f32_16x16x32_bf16(
                false, aC, false, bC[f], (short)0, acc[f], false, false);
        }
        if (more) {
            aC = aN;
#pragma unroll
            for (int f = 0; f < 4; ++f) bC[f] = bN[f];
        }
    }

    // C/D layout: lanes 0-15 -> rows mBase+v..; lanes 16-31 -> rows mBase+8+v
    const int rBase = mBase + ((lane >> 4) << 3);
#pragma unroll
    for (int f = 0; f < 4; ++f) {
        const int col = nBase + f * 16 + (lane & 15);
        if (SPLIT) {
            if (col < D_INNER) {
#pragma unroll
                for (int v = 0; v < 8; ++v)
                    C0[(size_t)(rBase + v) * D_INNER + col] = acc[f][v];
            } else {
                const int c2 = col - D_INNER;
#pragma unroll
                for (int v = 0; v < 8; ++v) {
                    float zv = acc[f][v];
                    C1[(size_t)(rBase + v) * D_INNER + c2] = zv / (1.0f + __expf(-zv));
                }
            }
        } else {
#pragma unroll
            for (int v = 0; v < 8; ++v)
                C0[(size_t)(rBase + v) * LDC + col] = acc[f][v];
        }
    }
}

// ---------------------------------------------------------------------------
// Fused: depthwise 3x3 conv (+bias) -> SiLU -> * (0.25*sum_k Ds) -> * silu(z)
//        -> LayerNorm(gamma,beta) -> bf16
// One block per pixel, 128 threads, 3 channels per thread (384 total).
// ---------------------------------------------------------------------------
__global__ __launch_bounds__(128)
void conv_ln_fuse(const float* __restrict__ xh, const float* __restrict__ zs,
                  const float* __restrict__ conv_w, const float* __restrict__ conv_b,
                  const float* __restrict__ Ds, const float* __restrict__ gamma,
                  const float* __restrict__ beta, __bf16* __restrict__ ybf) {
    __shared__ float s_sum[128];
    __shared__ float s_sq[128];

    const int pix = blockIdx.x;
    const int b   = pix >> 12;
    const int rem = pix & 4095;
    const int h   = rem >> 6;
    const int w   = rem & 63;
    const int tid = threadIdx.x;

    float vloc[3];
    float lsum = 0.0f, lsq = 0.0f;

#pragma unroll
    for (int t = 0; t < 3; ++t) {
        const int c = tid + t * 128;
        float acc = conv_b[c];
#pragma unroll
        for (int kh = 0; kh < 3; ++kh) {
            const int hh = h + kh - 1;
            if (hh < 0 || hh >= HH) continue;
#pragma unroll
            for (int kw = 0; kw < 3; ++kw) {
                const int ww = w + kw - 1;
                if (ww < 0 || ww >= WW) continue;
                acc += xh[(size_t)(((b * HH + hh) * WW) + ww) * D_INNER + c] *
                       conv_w[(kh * 3 + kw) * D_INNER + c];
            }
        }
        const float s = acc / (1.0f + __expf(-acc));                 // SiLU
        const float dsc = 0.25f * (Ds[4 * c] + Ds[4 * c + 1] +
                                   Ds[4 * c + 2] + Ds[4 * c + 3]);   // merged scans
        const float val = s * dsc * zs[(size_t)pix * D_INNER + c];   // * silu(z)
        vloc[t] = val;
        lsum += val;
        lsq  += val * val;
    }

    s_sum[tid] = lsum;
    s_sq[tid]  = lsq;
    __syncthreads();
#pragma unroll
    for (int st = 64; st > 0; st >>= 1) {
        if (tid < st) { s_sum[tid] += s_sum[tid + st]; s_sq[tid] += s_sq[tid + st]; }
        __syncthreads();
    }
    const float mu   = s_sum[0] * (1.0f / (float)D_INNER);
    const float var  = s_sq[0] * (1.0f / (float)D_INNER) - mu * mu;
    const float rstd = rsqrtf(fmaxf(var, 0.0f) + 1e-5f);

#pragma unroll
    for (int t = 0; t < 3; ++t) {
        const int c = tid + t * 128;
        ybf[(size_t)pix * D_INNER + c] =
            f2bf((vloc[t] - mu) * rstd * gamma[c] + beta[c]);
    }
}

// ---------------------------------------------------------------------------
// Host launcher
// ---------------------------------------------------------------------------
extern "C" void kernel_launch(void* const* d_in, const int* in_sizes, int n_in,
                              void* d_out, int out_size, void* d_ws, size_t ws_size,
                              hipStream_t stream) {
    const float* x          = (const float*)d_in[0];
    const float* in_proj_w  = (const float*)d_in[1];   // (768, 192)
    const float* conv_w     = (const float*)d_in[2];   // (3,3,1,384)
    const float* conv_b     = (const float*)d_in[3];   // (384,)
    const float* Ds         = (const float*)d_in[8];   // (1536,)
    const float* ln_gamma   = (const float*)d_in[9];
    const float* ln_beta    = (const float*)d_in[10];
    const float* out_proj_w = (const float*)d_in[11];  // (192, 384)
    float* out = (float*)d_out;

    char* ws = (char*)d_ws;
    size_t off = 0;
    auto alloc = [&](size_t bytes) -> void* {
        void* p = ws + off;
        off = (off + bytes + 255) & ~(size_t)255;
        return p;
    };
    __bf16* xbf     = (__bf16*)alloc((size_t)M_TOTAL * D_MODEL * 2);  // 12.6 MB
    __bf16* wbf_in  = (__bf16*)alloc((size_t)768 * D_MODEL * 2);      // 0.3 MB
    __bf16* wbf_out = (__bf16*)alloc((size_t)D_MODEL * D_INNER * 2);  // 0.15 MB
    __bf16* ybf     = (__bf16*)alloc((size_t)M_TOTAL * D_INNER * 2);  // 25.2 MB
    float*  xh      = (float*)alloc((size_t)M_TOTAL * D_INNER * 4);   // 50.3 MB
    float*  zs      = (float*)alloc((size_t)M_TOTAL * D_INNER * 4);   // 50.3 MB

    // 1) bf16 conversions
    {
        int n = M_TOTAL * D_MODEL;
        cvt_f32_bf16<<<(n + 255) / 256, 256, 0, stream>>>(x, xbf, n);
        n = 768 * D_MODEL;
        cvt_f32_bf16<<<(n + 255) / 256, 256, 0, stream>>>(in_proj_w, wbf_in, n);
        n = D_MODEL * D_INNER;
        cvt_f32_bf16<<<(n + 255) / 256, 256, 0, stream>>>(out_proj_w, wbf_out, n);
    }

    // 2) in_proj GEMM: [32768 x 192] x [192 x 768] -> xh (raw) + zs (= silu(z))
    wmma_gemm_bt<D_MODEL, D_INNER, true>
        <<<dim3(M_TOTAL / 128, 768 / 64), 256, 0, stream>>>(xbf, wbf_in, xh, zs);

    // 3) dwconv + SiLU + Ds-scale + *silu(z) + LayerNorm -> bf16 y
    conv_ln_fuse<<<M_TOTAL, 128, 0, stream>>>(xh, zs, conv_w, conv_b, Ds,
                                              ln_gamma, ln_beta, ybf);

    // 4) out_proj GEMM: [32768 x 384] x [384 x 192] -> out (fp32)
    wmma_gemm_bt<D_INNER, D_MODEL, false>
        <<<dim3(M_TOTAL / 128, D_MODEL / 64), 256, 0, stream>>>(ybf, wbf_out, out, nullptr);
}
